// BatchedGAT_cat1Temporal_40862318854440
// MI455X (gfx1250) — matrix operand order repaired and checked by reading, not compiled, not run on hardware
//
#include <hip/hip_runtime.h>
#include <hip/hip_bf16.h>

// ---------------------------------------------------------------------------
// BatchedGAT (B=16,N=128,K=16,D_IN=512,D_OUT=512) for MI455X / gfx1250.
// Memory-bound (~80MB traffic, ~4.3 GFLOP): bf16 WMMA GEMMs with f32
// accumulate, gather-based attention (e = s_self[n] + s_nei[idx]), fused
// epilogues. Wave32 v_wmma_f32_16x16x32_bf16; 32x32 C tile per wave
// (4 accumulators, A/B fragments each reused twice -> 2 b128 loads per WMMA).
// ---------------------------------------------------------------------------

typedef __bf16 bf16_t;
typedef __attribute__((ext_vector_type(16))) __bf16 v16bf;
typedef __attribute__((ext_vector_type(8)))  float  v8f;

#define BB   16
#define NN   128
#define KK   16
#define DIN  512
#define DOUT 512
#define ROWS (BB * NN)        // 2048
#define CH   (4 * DOUT)       // 2048

union BFrag { uint4 q[2]; v16bf v; };

// ---------------------------------------------------------------------------
// 1) prep: x -> bf16 copy; s_self = x . Wa[0:512]; s_nei = x . Wa[512:1024]
// ---------------------------------------------------------------------------
__global__ __launch_bounds__(256) void prep_kernel(
    const float* __restrict__ x, const float* __restrict__ Wa,
    bf16_t* __restrict__ xbf, float* __restrict__ s_self, float* __restrict__ s_nei)
{
    const int bn  = blockIdx.x;
    const int tid = threadIdx.x;
    __shared__ float r0[256], r1[256];
    const float* row = x + (size_t)bn * DIN;
    float a0 = 0.f, a1 = 0.f;
    for (int d = tid; d < DIN; d += 256) {
        float v = row[d];
        xbf[(size_t)bn * DIN + d] = (bf16_t)v;
        a0 += v * Wa[d];
        a1 += v * Wa[DIN + d];
    }
    r0[tid] = a0; r1[tid] = a1;
    __syncthreads();
    for (int s = 128; s > 0; s >>= 1) {
        if (tid < s) { r0[tid] += r0[tid + s]; r1[tid] += r1[tid + s]; }
        __syncthreads();
    }
    if (tid == 0) { s_self[bn] = r0[0]; s_nei[bn] = r1[0]; }
}

// ---------------------------------------------------------------------------
// 2) convert projection weights to bf16
// ---------------------------------------------------------------------------
__global__ __launch_bounds__(256) void convw_kernel(
    const float* __restrict__ wx, const float* __restrict__ wn,
    bf16_t* __restrict__ wxbf, bf16_t* __restrict__ wnbf)
{
    int i = blockIdx.x * 256 + threadIdx.x;
    if (i < DOUT * DIN) {
        wxbf[i] = (bf16_t)wx[i];
        wnbf[i] = (bf16_t)wn[i];
    }
}

// ---------------------------------------------------------------------------
// 3) attention aggregate per branch: softmax_k(leaky(s_self + s_nei[idx])),
//    agg[bn,d] = sum_k attn_k * x[b, idx[n,k], d].  grid = (2048, 3)
// ---------------------------------------------------------------------------
__global__ __launch_bounds__(256) void agg_kernel(
    const bf16_t* __restrict__ xbf,
    const int* __restrict__ idx1, const int* __restrict__ idx2, const int* __restrict__ idx3,
    const float* __restrict__ s_self, const float* __restrict__ s_nei,
    bf16_t* __restrict__ aggbf)
{
    const int bn  = blockIdx.x;
    const int g   = blockIdx.y;
    const int b   = bn >> 7;
    const int n   = bn & (NN - 1);
    const int tid = threadIdx.x;
    const int* idx = (g == 0) ? idx1 : ((g == 1) ? idx2 : idx3);

    __shared__ float attn[KK];
    __shared__ int   nb[KK];

    if (tid < KK) {
        int j = idx[n * KK + tid];
        nb[tid] = j;
        float e = s_self[bn] + s_nei[b * NN + j];
        attn[tid] = (e > 0.f) ? e : 0.2f * e;   // LeakyReLU(0.2)
    }
    __syncthreads();
    if (tid == 0) {
        float mx = attn[0];
        for (int k = 1; k < KK; ++k) mx = fmaxf(mx, attn[k]);
        float ex[KK]; float sum = 0.f;
        for (int k = 0; k < KK; ++k) { ex[k] = expf(attn[k] - mx); sum += ex[k]; }
        float inv = 1.f / sum;
        for (int k = 0; k < KK; ++k) attn[k] = ex[k] * inv;
    }
    __syncthreads();

    float w[KK]; int ro[KK];
    #pragma unroll
    for (int k = 0; k < KK; ++k) { w[k] = attn[k]; ro[k] = (b * NN + nb[k]) * DIN; }

    bf16_t* out = aggbf + (size_t)g * ROWS * DIN + (size_t)bn * DIN;
    for (int d = tid; d < DIN; d += 256) {
        float acc = 0.f;
        #pragma unroll
        for (int k = 0; k < KK; ++k) acc += w[k] * (float)xbf[ro[k] + d];
        out[d] = (bf16_t)acc;
    }
}

// ---------------------------------------------------------------------------
// 4) GEMM: h[:, p*512 : (p+1)*512] = A_p (2048x512, bf16) @ W_p^T + bias_p
//    blockIdx.y = p in 0..3 (p=0: x@Wx^T, p=1..3: agg_g@Wn^T).
//    8 waves/block; each wave owns a 32x32 C tile: 2 A-frags x 2 B-frags ->
//    4 v_wmma_f32_16x16x32_bf16 per K-step, every fragment reused twice.
// ---------------------------------------------------------------------------
__global__ __launch_bounds__(256) void gemm_kernel(
    const bf16_t* __restrict__ xbf, const bf16_t* __restrict__ aggbf,
    const bf16_t* __restrict__ wxbf, const bf16_t* __restrict__ wnbf,
    const float* __restrict__ bias_x, const float* __restrict__ bias_n,
    float* __restrict__ h)
{
    const int p = blockIdx.y;                 // 0..3
    const bf16_t* A    = (p == 0) ? xbf  : aggbf + (size_t)(p - 1) * ROWS * DIN;
    const bf16_t* W    = (p == 0) ? wxbf : wnbf;
    const float*  bias = (p == 0) ? bias_x : bias_n;

    const int wave = threadIdx.x >> 5;
    const int lane = threadIdx.x & 31;
    const int t    = blockIdx.x * 8 + wave;   // tile id, 0..1023
    const int tm   = t & 63;                  // 64 M tiles of 32
    const int tn   = t >> 6;                  // 16 N tiles of 32
    const int lmod = lane & 15;
    const int lhi  = lane >> 4;

    const bf16_t* A0 = A + (size_t)(tm * 32 + lmod) * DIN;        // M rows 0..15
    const bf16_t* A1 = A + (size_t)(tm * 32 + 16 + lmod) * DIN;   // M rows 16..31
    const bf16_t* W0 = W + (size_t)(tn * 32 + lmod) * DIN;        // N cols 0..15
    const bf16_t* W1 = W + (size_t)(tn * 32 + 16 + lmod) * DIN;   // N cols 16..31

    v8f c00 = {}; v8f c01 = {}; v8f c10 = {}; v8f c11 = {};
    #pragma unroll 4
    for (int k0 = 0; k0 < DIN; k0 += 32) {
        BFrag a0, a1, b0, b1;
        // A 16x32 bf16 frag: lane covers K = k0+lhi*8+[0..7] and k0+16+lhi*8+[0..7]
        a0.q[0] = *(const uint4*)(A0 + k0 + lhi * 8);
        a0.q[1] = *(const uint4*)(A0 + k0 + 16 + lhi * 8);
        a1.q[0] = *(const uint4*)(A1 + k0 + lhi * 8);
        a1.q[1] = *(const uint4*)(A1 + k0 + 16 + lhi * 8);
        // B 32x16 frag (B[k,n]=W[n,k]): lane covers K = k0+lhi*8+[0..7] and k0+16+lhi*8+[0..7]
        b0.q[0] = *(const uint4*)(W0 + k0 + lhi * 8);
        b0.q[1] = *(const uint4*)(W0 + k0 + 16 + lhi * 8);
        b1.q[0] = *(const uint4*)(W1 + k0 + lhi * 8);
        b1.q[1] = *(const uint4*)(W1 + k0 + 16 + lhi * 8);
        c00 = __builtin_amdgcn_wmma_f32_16x16x32_bf16(false, a0.v, false, b0.v,
                                                      (short)0, c00, false, false);
        c01 = __builtin_amdgcn_wmma_f32_16x16x32_bf16(false, a0.v, false, b1.v,
                                                      (short)0, c01, false, false);
        c10 = __builtin_amdgcn_wmma_f32_16x16x32_bf16(false, a1.v, false, b0.v,
                                                      (short)0, c10, false, false);
        c11 = __builtin_amdgcn_wmma_f32_16x16x32_bf16(false, a1.v, false, b1.v,
                                                      (short)0, c11, false, false);
    }

    // C/D layout: VGPR r -> row r + 8*lhi, col lmod
    const int colbase = p * DOUT + tn * 32;
    const float bi0 = bias[tn * 32 + lmod];
    const float bi1 = bias[tn * 32 + 16 + lmod];
    #pragma unroll
    for (int r = 0; r < 8; ++r) {
        int row0 = tm * 32 + r + 8 * lhi;
        int row1 = row0 + 16;
        h[(size_t)row0 * CH + colbase + lmod]      = c00[r] + bi0;
        h[(size_t)row0 * CH + colbase + 16 + lmod] = c01[r] + bi1;
        h[(size_t)row1 * CH + colbase + lmod]      = c10[r] + bi0;
        h[(size_t)row1 * CH + colbase + 16 + lmod] = c11[r] + bi1;
    }
}

// ---------------------------------------------------------------------------
// 5) fused row L2-normalize + ReLU (in place on h); one block per row
// ---------------------------------------------------------------------------
__global__ __launch_bounds__(256) void l2relu_kernel(float* __restrict__ h)
{
    const int bn  = blockIdx.x;
    const int tid = threadIdx.x;
    __shared__ float red[256];
    float* row = h + (size_t)bn * CH;
    float ss = 0.f;
    for (int c = tid; c < CH; c += 256) { float v = row[c]; ss += v * v; }
    red[tid] = ss;
    __syncthreads();
    for (int s = 128; s > 0; s >>= 1) {
        if (tid < s) red[tid] += red[tid + s];
        __syncthreads();
    }
    float scale = 1.f / fmaxf(sqrtf(red[0]), 1e-12f);
    for (int c = tid; c < CH; c += 256) {
        float v = row[c] * scale;
        row[c] = (v > 0.f) ? v : 0.f;
    }
}

// ---------------------------------------------------------------------------
// 6) per-channel mean / rsqrt(var+eps); one thread per channel (grid 8x256)
// ---------------------------------------------------------------------------
__global__ __launch_bounds__(256) void stats_kernel(
    const float* __restrict__ h, float* __restrict__ mu, float* __restrict__ rsig)
{
    const int ch = blockIdx.x * 256 + threadIdx.x;
    float s = 0.f, sq = 0.f;
    for (int r = 0; r < ROWS; ++r) {
        float v = h[(size_t)r * CH + ch];
        s += v; sq += v * v;
    }
    const float inv = 1.f / (float)ROWS;
    float m   = s * inv;
    float var = sq * inv - m * m;
    mu[ch]   = m;
    rsig[ch] = rsqrtf(var + 1e-5f);
}

// ---------------------------------------------------------------------------
// 7) apply batchnorm affine -> d_out
// ---------------------------------------------------------------------------
__global__ __launch_bounds__(256) void bnapply_kernel(
    const float* __restrict__ h, const float* __restrict__ mu, const float* __restrict__ rsig,
    const float* __restrict__ gamma, const float* __restrict__ beta, float* __restrict__ out)
{
    size_t i = (size_t)blockIdx.x * 256 + threadIdx.x;
    if (i < (size_t)ROWS * CH) {
        int ch = (int)(i & (CH - 1));
        out[i] = (h[i] - mu[ch]) * rsig[ch] * gamma[ch] + beta[ch];
    }
}

// ---------------------------------------------------------------------------
extern "C" void kernel_launch(void* const* d_in, const int* in_sizes, int n_in,
                              void* d_out, int out_size, void* d_ws, size_t ws_size,
                              hipStream_t stream) {
    const float* x     = (const float*)d_in[0];
    const int*   idx1  = (const int*)  d_in[1];
    const int*   idx2  = (const int*)  d_in[2];
    const int*   idx3  = (const int*)  d_in[3];
    const float* Wx_w  = (const float*)d_in[4];
    const float* Wx_b  = (const float*)d_in[5];
    const float* Wn_w  = (const float*)d_in[6];
    const float* Wn_b  = (const float*)d_in[7];
    const float* Wa    = (const float*)d_in[8];
    const float* gamma = (const float*)d_in[9];
    const float* beta  = (const float*)d_in[10];
    float* out = (float*)d_out;

    // workspace carve-out (all sizes multiples of 256B)
    char* w = (char*)d_ws;
    bf16_t* xbf   = (bf16_t*)w; w += (size_t)ROWS * DIN * 2;        // 2 MB
    bf16_t* wxbf  = (bf16_t*)w; w += (size_t)DOUT * DIN * 2;        // 0.5 MB
    bf16_t* wnbf  = (bf16_t*)w; w += (size_t)DOUT * DIN * 2;        // 0.5 MB
    bf16_t* aggbf = (bf16_t*)w; w += (size_t)3 * ROWS * DIN * 2;    // 6 MB
    float* s_self = (float*)w;  w += (size_t)ROWS * 4;
    float* s_nei  = (float*)w;  w += (size_t)ROWS * 4;
    float* h      = (float*)w;  w += (size_t)ROWS * CH * 4;         // 16.8 MB
    float* mu     = (float*)w;  w += (size_t)CH * 4;
    float* rsig   = (float*)w;  w += (size_t)CH * 4;

    prep_kernel<<<ROWS, 256, 0, stream>>>(x, Wa, xbf, s_self, s_nei);
    convw_kernel<<<(DOUT * DIN + 255) / 256, 256, 0, stream>>>(Wx_w, Wn_w, wxbf, wnbf);
    agg_kernel<<<dim3(ROWS, 3), 256, 0, stream>>>(xbf, idx1, idx2, idx3, s_self, s_nei, aggbf);
    gemm_kernel<<<dim3(128, 4), 256, 0, stream>>>(xbf, aggbf, wxbf, wnbf, Wx_b, Wn_b, h);
    l2relu_kernel<<<ROWS, 256, 0, stream>>>(h);
    stats_kernel<<<CH / 256, 256, 0, stream>>>(h, mu, rsig);
    bnapply_kernel<<<(ROWS * CH + 255) / 256, 256, 0, stream>>>(h, mu, rsig, gamma, beta, out);
}
